// ConvBnA_int_31628139168649
// MI455X (gfx1250) — 8-bit matrix operand order repaired and checked by reading, compiled, not run on hardware
//
#include <hip/hip_runtime.h>
#include <stdint.h>

typedef int v8i __attribute__((ext_vector_type(8)));

#define CIN   128
#define COUT  256
#define HH    56
#define WW    56
#define NBAT  32
#define HW    (HH*WW)
#define PTOT  (NBAT*HW)
#define XPK_BYTES ((size_t)NBAT*HW*CIN)
#define LDS_PITCH 80
#define TILE_P  256
#define KSTEPS  18

__global__ __launch_bounds__(256) void pack_x_kernel(const int* __restrict__ x,
                                                     int8_t* __restrict__ xpk)
{
    int idx = blockIdx.x * 256 + threadIdx.x;
    if (idx >= PTOT * (CIN / 4)) return;
    int s   = idx >> 5;
    int c4  = (idx & 31) << 2;
    int n   = s / HW;
    int rem = s - n * HW;
    const int* px = x + ((size_t)n * CIN + c4) * HW + rem;
    uint32_t b0 = (uint32_t)(uint8_t)(int8_t)px[0 * HW];
    uint32_t b1 = (uint32_t)(uint8_t)(int8_t)px[1 * HW];
    uint32_t b2 = (uint32_t)(uint8_t)(int8_t)px[2 * HW];
    uint32_t b3 = (uint32_t)(uint8_t)(int8_t)px[3 * HW];
    *(uint32_t*)(xpk + (size_t)s * CIN + c4) = b0 | (b1 << 8) | (b2 << 16) | (b3 << 24);
}

__global__ __launch_bounds__(256) void pack_w_kernel(const int* __restrict__ w,
                                                     int8_t* __restrict__ wpk)
{
    int idx = blockIdx.x * 256 + threadIdx.x;
    if (idx >= 9 * COUT * (CIN / 4)) return;
    int khw  = idx >> 13;
    int r    = idx & 8191;
    int cout = r >> 5;
    int c4   = (r & 31) << 2;
    const int* pw = w + ((size_t)cout * CIN + c4) * 9 + khw;
    uint32_t b0 = (uint32_t)(uint8_t)(int8_t)pw[0 * 9];
    uint32_t b1 = (uint32_t)(uint8_t)(int8_t)pw[1 * 9];
    uint32_t b2 = (uint32_t)(uint8_t)(int8_t)pw[2 * 9];
    uint32_t b3 = (uint32_t)(uint8_t)(int8_t)pw[3 * 9];
    *(uint32_t*)(wpk + ((size_t)khw * COUT + cout) * CIN + c4) =
        b0 | (b1 << 8) | (b2 << 16) | (b3 << 24);
}

__global__ __launch_bounds__(256) void conv_wmma_iu8_kernel(
    const int8_t* __restrict__ xpk, const int8_t* __restrict__ wpk,
    const int* __restrict__ nshift, const int* __restrict__ tbias,
    const int* __restrict__ actmin, const int* __restrict__ actmax,
    float* __restrict__ out)
{
    __shared__ __align__(16) int8_t Bs[2][TILE_P * LDS_PITCH];

    const int tid  = threadIdx.x;
    const int lane = tid & 31;
    const int wave = tid >> 5;
    const int wm   = wave & 3;
    const int wn   = wave >> 2;

    const int coutBase = blockIdx.y * 64;
    const int pixBase  = blockIdx.x * TILE_P;

    const int spix = tid;
    const int sp   = pixBase + spix;
    const int sn   = sp / HW;
    const int srem = sp - sn * HW;
    const int soh  = srem / WW;
    const int sow  = srem - soh * WW;

    const int arow = coutBase + wm * 16 + (lane & 15);
    const int ako  = (lane < 16) ? 0 : 8;
    const int bcol = lane & 15;
    const int bko  = (lane < 16) ? 0 : 8;
    const int bbase0 = (wn * 128 + bcol) * LDS_PITCH + bko;

    v8i acc[8] = {};

    auto stage = [&](int it, int buf) {
        const int khw = it >> 1;
        const int cc  = (it & 1) << 6;
        const int kh  = khw / 3;
        const int kw  = khw - kh * 3;
        const int ih  = soh + kh - 1;
        const int iw  = sow + kw - 1;
        const bool valid = ((unsigned)ih < HH) && ((unsigned)iw < WW);
        uint4 d0 = {0,0,0,0}, d1 = {0,0,0,0}, d2 = {0,0,0,0}, d3 = {0,0,0,0};
        if (valid) {
            const int8_t* src = xpk + (((size_t)sn * HH + ih) * WW + iw) * CIN + cc;
            d0 = *(const uint4*)(src);
            d1 = *(const uint4*)(src + 16);
            d2 = *(const uint4*)(src + 32);
            d3 = *(const uint4*)(src + 48);
        }
        int8_t* dst = Bs[buf] + spix * LDS_PITCH;
        *(uint4*)(dst)      = d0;
        *(uint4*)(dst + 16) = d1;
        *(uint4*)(dst + 32) = d2;
        *(uint4*)(dst + 48) = d3;
    };

    stage(0, 0);

    for (int it = 0; it < KSTEPS; ++it) {
        const int buf = it & 1;
        __syncthreads();
        if (it + 1 < KSTEPS) stage(it + 1, buf ^ 1);

        const int khw = it >> 1;
        const int cc  = (it & 1) << 6;
        const int8_t* wsrc = wpk + ((size_t)(khw * COUT + arow)) * CIN + cc + ako;
        int2 a01 = *(const int2*)(wsrc);
        int2 a23 = *(const int2*)(wsrc + 16);
        int2 a45 = *(const int2*)(wsrc + 32);
        int2 a67 = *(const int2*)(wsrc + 48);
        v8i A;
        A[0] = a01.x; A[1] = a01.y; A[2] = a23.x; A[3] = a23.y;
        A[4] = a45.x; A[5] = a45.y; A[6] = a67.x; A[7] = a67.y;

        #pragma unroll
        for (int nt = 0; nt < 8; ++nt) {
            const int8_t* bp = Bs[buf] + bbase0 + nt * 16 * LDS_PITCH;
            int2 b0 = *(const int2*)(bp);
            int2 b1 = *(const int2*)(bp + 16);
            int2 b2 = *(const int2*)(bp + 32);
            int2 b3 = *(const int2*)(bp + 48);
            v8i B;
            B[0] = b0.x; B[1] = b0.y; B[2] = b1.x; B[3] = b1.y;
            B[4] = b2.x; B[5] = b2.y; B[6] = b3.x; B[7] = b3.y;
            acc[nt] = __builtin_amdgcn_wmma_i32_16x16x64_iu8(
                 true, A,  true, B, acc[nt],
                 false,  false);
        }
    }

    const int amin  = actmin[0];
    const int amax  = actmax[0];
    const int rowOff = (lane < 16) ? 0 : 8;
    const int co0    = coutBase + wm * 16 + rowOff;

    int bias8[8], sh8[8];
    #pragma unroll
    for (int r = 0; r < 8; ++r) {
        bias8[r] = tbias[co0 + r];
        sh8[r]   = -nshift[co0 + r];
    }

    #pragma unroll
    for (int nt = 0; nt < 8; ++nt) {
        const int pix  = pixBase + wn * 128 + nt * 16 + bcol;
        const int pn   = pix / HW;
        const int prem = pix - pn * HW;
        float* obase = out + (size_t)pn * (COUT * HW) + prem;
        #pragma unroll
        for (int r = 0; r < 8; ++r) {
            int v = (acc[nt][r] + bias8[r]) >> sh8[r];
            v = v < amin ? amin : (v > amax ? amax : v);
            obase[(size_t)(co0 + r) * HW] = (float)v;
        }
    }
}

extern "C" void kernel_launch(void* const* d_in, const int* in_sizes, int n_in,
                              void* d_out, int out_size, void* d_ws, size_t ws_size,
                              hipStream_t stream)
{
    const int* x    = (const int*)d_in[0];
    const int* w    = (const int*)d_in[1];
    const int* nsh  = (const int*)d_in[2];
    const int* tb   = (const int*)d_in[3];
    const int* amin = (const int*)d_in[4];
    const int* amax = (const int*)d_in[5];

    int8_t* xpk = (int8_t*)d_ws;
    int8_t* wpk = (int8_t*)d_ws + XPK_BYTES;

    pack_x_kernel<<<(PTOT * (CIN / 4) + 255) / 256, 256, 0, stream>>>(x, xpk);
    pack_w_kernel<<<(9 * COUT * (CIN / 4) + 255) / 256, 256, 0, stream>>>(w, wpk);

    dim3 grid(PTOT / TILE_P, COUT / 64);
    conv_wmma_iu8_kernel<<<grid, 256, 0, stream>>>(xpk, wpk, nsh, tb, amin, amax,
                                                   (float*)d_out);
}
